// GATSBY_75402445848771
// MI455X (gfx1250) — compile-verified, matrix operand order checked by reading, not run on hardware
//
#include <hip/hip_runtime.h>
#include <hip/hip_bf16.h>
#include <math.h>

typedef __attribute__((ext_vector_type(16))) _Float16 v16h;
typedef __attribute__((ext_vector_type(8)))  float    v8f;

#define NEG_SLOPE 0.2f
#define GAT_EPS   1e-16f

// ---- monotone float <-> uint mapping so float max == uint max -----------
__device__ __forceinline__ unsigned f2key(float f) {
    unsigned b = __float_as_uint(f);
    return (b & 0x80000000u) ? ~b : (b | 0x80000000u);
}
__device__ __forceinline__ float key2f(unsigned k) {
    return __uint_as_float((k & 0x80000000u) ? (k ^ 0x80000000u) : ~k);
}

// ---- WMMA 16-bit fragment index map: element i (0..15) of lane l's v16h
// covers K index  16*(i/8) + 8*(l/16) + 2*((i/2)%4) + (i%2)
__device__ __forceinline__ int kmap(int i, int l) {
    return 16 * (i >> 3) + 8 * (l >> 4) + 2 * ((i >> 1) & 3) + (i & 1);
}

// ---------------------------------------------------------------------------
// Zero-fill (grid-stride, float4)
// ---------------------------------------------------------------------------
__global__ void zero_f32(float4* p, long long n4) {
    long long i  = (long long)blockIdx.x * blockDim.x + threadIdx.x;
    long long st = (long long)gridDim.x * blockDim.x;
    float4 z = {0.f, 0.f, 0.f, 0.f};
    for (; i < n4; i += st) p[i] = z;
}

// ---------------------------------------------------------------------------
// Pre-pack weight matrix W[KDIM x NCOLS] (f32 row-major) into f16 WMMA
// B-fragments: pack[((ks*NT + t)*32 + lane)] is lane's v16h for k-step ks,
// n-tile t. 16KB per matrix -> L0/L2 resident, broadcast to all waves.
// ---------------------------------------------------------------------------
template <int KDIM, int NCOLS>
__global__ void pack_w(const float* __restrict__ W, v16h* __restrict__ P) {
    constexpr int NT = NCOLS / 16, KS = KDIM / 32;
    int idx = blockIdx.x * blockDim.x + threadIdx.x;
    if (idx >= KS * NT * 32) return;
    int l = idx & 31, t = (idx >> 5) % NT, ks = idx / (32 * NT);
    v16h v;
#pragma unroll
    for (int i = 0; i < 16; ++i)
        v[i] = (_Float16)W[(ks * 32 + kmap(i, l)) * NCOLS + t * 16 + (l & 15)];
    P[(ks * NT + t) * 32 + l] = v;
}

// ---------------------------------------------------------------------------
// WMMA GEMM: C[M x NCOLS] = A[M x KDIM] * Bpack  (f16 multiply, f32 acc).
// One wave per 16-row strip; A loaded as 4x b128 per k-step; B fragments
// as single 32B vector loads from the pre-packed buffer.
// ---------------------------------------------------------------------------
template <int KDIM, int NCOLS>
__global__ void gemm_wmma(const float* __restrict__ A,
                          const v16h* __restrict__ Bpack,
                          float* __restrict__ C, int M) {
    int wave = (int)(((long long)blockIdx.x * blockDim.x + threadIdx.x) >> 5);
    int l    = threadIdx.x & 31;
    int mTiles = (M + 15) >> 4;
    if (wave >= mTiles) return;            // wave-uniform: EXEC stays all-1s

    int m0   = wave << 4;
    int rowA = m0 + (l & 15);
    int rA   = rowA < M ? rowA : (M - 1);  // clamp tail rows (stores masked)

    constexpr int NT = NCOLS / 16;
    v8f zero = {0.f, 0.f, 0.f, 0.f, 0.f, 0.f, 0.f, 0.f};
    v8f acc[NT];
#pragma unroll
    for (int t = 0; t < NT; ++t) acc[t] = zero;

    const float* arow = A + (long long)rA * KDIM + 8 * (l >> 4);

#pragma unroll
    for (int k0 = 0; k0 < KDIM; k0 += 32) {
        // lane's A elements: two contiguous 8-float runs at k0+8*(l/16)[+16]
        const float4* ap = (const float4*)(arow + k0);
        float4 x0 = ap[0], x1 = ap[1], x2 = ap[4], x3 = ap[5];
        v16h a;
        a[0]  = (_Float16)x0.x; a[1]  = (_Float16)x0.y;
        a[2]  = (_Float16)x0.z; a[3]  = (_Float16)x0.w;
        a[4]  = (_Float16)x1.x; a[5]  = (_Float16)x1.y;
        a[6]  = (_Float16)x1.z; a[7]  = (_Float16)x1.w;
        a[8]  = (_Float16)x2.x; a[9]  = (_Float16)x2.y;
        a[10] = (_Float16)x2.z; a[11] = (_Float16)x2.w;
        a[12] = (_Float16)x3.x; a[13] = (_Float16)x3.y;
        a[14] = (_Float16)x3.z; a[15] = (_Float16)x3.w;
#pragma unroll
        for (int t = 0; t < NT; ++t) {
            v16h b = Bpack[((k0 >> 5) * NT + t) * 32 + l];
            acc[t] = __builtin_amdgcn_wmma_f32_16x16x32_f16(
                false, a, false, b, (short)0, acc[t], false, false);
        }
    }

    int rbase = m0 + 8 * (l >> 4);
    int col   = l & 15;
#pragma unroll
    for (int t = 0; t < NT; ++t) {
#pragma unroll
        for (int j = 0; j < 8; ++j) {
            int r = rbase + j;
            if (r < M) C[(long long)r * NCOLS + t * 16 + col] = acc[t][j];
        }
    }
}

// ---------------------------------------------------------------------------
// Layer-1 per-node attention logits: h1 is [N,8,8]
// ---------------------------------------------------------------------------
__global__ void logits1(const float* __restrict__ h1,
                        const float* __restrict__ att_s,
                        const float* __restrict__ att_d,
                        float* asrc, float* adst, int N) {
    int idx = blockIdx.x * blockDim.x + threadIdx.x;  // n*8 + h
    if (idx >= N * 8) return;
    int h = idx & 7;
    const float4* row = (const float4*)(h1 + (long long)idx * 8);  // n*64+h*8
    const float4* s4  = (const float4*)(att_s + h * 8);
    const float4* d4  = (const float4*)(att_d + h * 8);
    float4 r0 = row[0], r1 = row[1];
    float4 a0 = s4[0],  a1 = s4[1];
    float4 b0 = d4[0],  b1 = d4[1];
    float s = r0.x * a0.x + r0.y * a0.y + r0.z * a0.z + r0.w * a0.w
            + r1.x * a1.x + r1.y * a1.y + r1.z * a1.z + r1.w * a1.w;
    float d = r0.x * b0.x + r0.y * b0.y + r0.z * b0.z + r0.w * b0.w
            + r1.x * b1.x + r1.y * b1.y + r1.z * b1.z + r1.w * b1.w;
    asrc[idx] = s;
    adst[idx] = d;
}

// ---------------------------------------------------------------------------
// Layer-2 per-node attention logits: h2 is [N,128], single head
// ---------------------------------------------------------------------------
__global__ void logits2(const float* __restrict__ h2,
                        const float* __restrict__ att_s,
                        const float* __restrict__ att_d,
                        float* asrc, float* adst, int N) {
    int n = blockIdx.x * blockDim.x + threadIdx.x;
    if (n >= N) return;
    const float4* row = (const float4*)(h2 + (long long)n * 128);
    const float4* s4  = (const float4*)att_s;
    const float4* d4  = (const float4*)att_d;
    float s = 0.f, d = 0.f;
#pragma unroll 8
    for (int f = 0; f < 32; ++f) {
        float4 v = row[f], a = s4[f], b = d4[f];
        s += v.x * a.x + v.y * a.y + v.z * a.z + v.w * a.w;
        d += v.x * b.x + v.y * b.y + v.z * b.z + v.w * b.w;
    }
    asrc[n] = s;
    adst[n] = d;
}

// ---------------------------------------------------------------------------
// Edge pass 1: segment-max of LeakyReLU logits via uint-keyed atomics.
// One thread per edge; H a template param so H=8 vectorizes its gathers.
// Edges [0,E) come from edge_index; [E,EE) are self loops.
// ---------------------------------------------------------------------------
template <int H>
__global__ void edge_max(const long long* __restrict__ ei, long long E, long long EE,
                         const float* __restrict__ asrc, const float* __restrict__ adst,
                         unsigned* amaxk) {
    long long e = (long long)blockIdx.x * blockDim.x + threadIdx.x;
    if (e >= EE) return;
    long long s = (e < E) ? ei[e]     : (e - E);
    long long d = (e < E) ? ei[E + e] : (e - E);
#pragma unroll
    for (int h = 0; h < H; ++h) {
        float a = asrc[s * H + h] + adst[d * H + h];
        a = (a >= 0.f) ? a : NEG_SLOPE * a;
        atomicMax(&amaxk[d * H + h], f2key(a));
    }
}

// ---------------------------------------------------------------------------
// Layer-1 edge pass 2: one wave per edge. Lane l covers channels 2l,2l+1
// (head h = l/4) -> b64 vector gather. Accumulates exp-weighted messages
// and the softmax denominator in the same pass (normalized per-node later).
// ---------------------------------------------------------------------------
__global__ void edge_agg1(const long long* __restrict__ ei, long long E, long long EE,
                          const float* __restrict__ asrc, const float* __restrict__ adst,
                          const unsigned* __restrict__ amaxk, float* denom,
                          const float* __restrict__ h1, float* agg) {
    long long wave = ((long long)blockIdx.x * blockDim.x + threadIdx.x) >> 5;
    int l = threadIdx.x & 31;
    if (wave >= EE) return;
    long long e = wave;
    long long s = (e < E) ? ei[e]     : (e - E);
    long long d = (e < E) ? ei[E + e] : (e - E);
    int h = l >> 2;
    float a = asrc[s * 8 + h] + adst[d * 8 + h];
    a = (a >= 0.f) ? a : NEG_SLOPE * a;
    float ea = __expf(a - key2f(amaxk[d * 8 + h]));
    if ((l & 3) == 0) atomicAdd(&denom[d * 8 + h], ea);
    int c0 = l * 2;
    float2 hv = *(const float2*)(h1 + s * 64 + c0);
    atomicAdd(&agg[d * 64 + c0],     ea * hv.x);
    atomicAdd(&agg[d * 64 + c0 + 1], ea * hv.y);
}

// ---------------------------------------------------------------------------
// Layer-2 edge pass 2: one wave per edge, lane l covers channels 4l..4l+3
// -> b128 vector gather (contiguous 512B per wave).
// ---------------------------------------------------------------------------
__global__ void edge_agg2(const long long* __restrict__ ei, long long E, long long EE,
                          const float* __restrict__ asrc, const float* __restrict__ adst,
                          const unsigned* __restrict__ amaxk, float* denom,
                          const float* __restrict__ h2, float* out) {
    long long wave = ((long long)blockIdx.x * blockDim.x + threadIdx.x) >> 5;
    int l = threadIdx.x & 31;
    if (wave >= EE) return;
    long long e = wave;
    long long s = (e < E) ? ei[e]     : (e - E);
    long long d = (e < E) ? ei[E + e] : (e - E);
    float a = asrc[s] + adst[d];
    a = (a >= 0.f) ? a : NEG_SLOPE * a;
    float ea = __expf(a - key2f(amaxk[d]));
    if (l == 0) atomicAdd(&denom[d], ea);
    int c0 = l * 4;
    float4 hv = *(const float4*)(h2 + s * 128 + c0);
    float* dst = out + d * 128 + c0;
    atomicAdd(dst + 0, ea * hv.x);
    atomicAdd(dst + 1, ea * hv.y);
    atomicAdd(dst + 2, ea * hv.z);
    atomicAdd(dst + 3, ea * hv.w);
}

// ---------------------------------------------------------------------------
// Layer-1 finalize: normalize, +bias, ELU  (in place on agg -> h1act)
// ---------------------------------------------------------------------------
__global__ void finalize1(float* agg, const float* __restrict__ denom,
                          const float* __restrict__ bias, int N) {
    int idx = blockIdx.x * blockDim.x + threadIdx.x;  // n*64 + c
    if (idx >= N * 64) return;
    int n = idx >> 6, c = idx & 63, h = c >> 3;
    float v = agg[idx] / (denom[n * 8 + h] + GAT_EPS) + bias[c];
    agg[idx] = (v > 0.f) ? v : (__expf(v) - 1.f);     // ELU(alpha=1)
}

// ---------------------------------------------------------------------------
// Layer-2 finalize: normalize + bias (in place on d_out)
// ---------------------------------------------------------------------------
__global__ void finalize2(float* out, const float* __restrict__ denom,
                          const float* __restrict__ bias, int N) {
    int idx = blockIdx.x * blockDim.x + threadIdx.x;  // n*128 + f
    if (idx >= N * 128) return;
    int n = idx >> 7, f = idx & 127;
    out[idx] = out[idx] / (denom[n] + GAT_EPS) + bias[f];
}

// ---------------------------------------------------------------------------
extern "C" void kernel_launch(void* const* d_in, const int* in_sizes, int n_in,
                              void* d_out, int out_size, void* d_ws, size_t ws_size,
                              hipStream_t stream) {
    const float*     x     = (const float*)d_in[0];
    const long long* ei    = (const long long*)d_in[1];   // int64 [2,E]
    const float*     W1    = (const float*)d_in[2];
    const float*     as1   = (const float*)d_in[3];
    const float*     ad1   = (const float*)d_in[4];
    const float*     b1    = (const float*)d_in[5];
    const float*     W2    = (const float*)d_in[6];
    const float*     as2   = (const float*)d_in[7];
    const float*     ad2   = (const float*)d_in[8];
    const float*     b2    = (const float*)d_in[9];
    float*           out   = (float*)d_out;

    const int       N  = in_sizes[0] / 128;
    const long long E  = (long long)in_sizes[1] / 2;
    const long long EE = E + N;                            // + self loops

    // ---- workspace layout (floats): N*292 + packed weights ----
    float* ws = (float*)d_ws;
    long long N64 = (long long)N * 64;
    float*    h1    = ws;                           // N*64
    float*    agg1  = h1 + N64;                     // N*64 (becomes h1act)
    float*    asrc1 = agg1 + N64;                   // N*8
    float*    adst1 = asrc1 + 8LL * N;              // N*8
    unsigned* amax1 = (unsigned*)(adst1 + 8LL * N); // N*8
    float*    den1  = (float*)(amax1 + 8LL * N);    // N*8
    float*    h2    = den1 + 8LL * N;               // N*128
    float*    asrc2 = h2 + 128LL * N;               // N
    float*    adst2 = asrc2 + N;                    // N
    unsigned* amax2 = (unsigned*)(adst2 + N);       // N
    float*    den2  = (float*)(amax2 + N);          // N
    // packed weights: align to 64B, 8192 halfs (16KB) each
    uintptr_t pbase = ((uintptr_t)(den2 + N) + 63) & ~(uintptr_t)63;
    v16h* pack1 = (v16h*)pbase;                     // 512 v16h = 16KB
    v16h* pack2 = pack1 + 512;                      // 512 v16h = 16KB

    const int TB = 256;

    // 0) zero accumulators/keys (bit pattern 0 == minimal max-key) + out
    zero_f32<<<2048, TB, 0, stream>>>((float4*)ws, 292LL * N / 4);
    zero_f32<<<2048, TB, 0, stream>>>((float4*)out, 128LL * N / 4);

    // 1) pre-pack weights into WMMA B-fragment layout (f16)
    pack_w<128, 64><<<2, TB, 0, stream>>>(W1, pack1);   // 4*4*32 = 512 thr
    pack_w<64, 128><<<2, TB, 0, stream>>>(W2, pack2);   // 2*8*32 = 512 thr

    int mTiles = (N + 15) / 16;
    int gemmBlocks = (mTiles * 32 + TB - 1) / TB;

    // ---- layer 1 ----
    gemm_wmma<128, 64><<<gemmBlocks, TB, 0, stream>>>(x, pack1, h1, N);
    logits1<<<(N * 8 + TB - 1) / TB, TB, 0, stream>>>(h1, as1, ad1, asrc1, adst1, N);
    edge_max<8><<<(int)((EE + TB - 1) / TB), TB, 0, stream>>>(ei, E, EE, asrc1, adst1, amax1);
    edge_agg1<<<(int)((EE * 32 + TB - 1) / TB), TB, 0, stream>>>(
        ei, E, EE, asrc1, adst1, amax1, den1, h1, agg1);
    finalize1<<<(N * 64 + TB - 1) / TB, TB, 0, stream>>>(agg1, den1, b1, N);

    // ---- layer 2 ----
    gemm_wmma<64, 128><<<gemmBlocks, TB, 0, stream>>>(agg1, pack2, h2, N);
    logits2<<<(N + TB - 1) / TB, TB, 0, stream>>>(h2, as2, ad2, asrc2, adst2, N);
    edge_max<1><<<(int)((EE + TB - 1) / TB), TB, 0, stream>>>(ei, E, EE, asrc2, adst2, amax2);
    edge_agg2<<<(int)((EE * 32 + TB - 1) / TB), TB, 0, stream>>>(
        ei, E, EE, asrc2, adst2, amax2, den2, h2, out);
    finalize2<<<(N * 128 + TB - 1) / TB, TB, 0, stream>>>(out, den2, b2, N);
}
